// CopyLSTMDecoder_944892805497
// MI455X (gfx1250) — matrix operand order repaired and checked by reading, not compiled, not run on hardware
//
#include <hip/hip_runtime.h>

#define B_   32
#define T_   64
#define L_   512
#define H_   512
#define E_   256
#define V_   32000
#define EXT_ 32100

typedef __bf16 bf16;
typedef __attribute__((ext_vector_type(16))) __bf16 v16bf;
typedef __attribute__((ext_vector_type(8)))  __bf16 v8bf;
typedef __attribute__((ext_vector_type(8)))  float  v8f;

__device__ __forceinline__ float sigf(float x) { return 1.0f / (1.0f + __expf(-x)); }

// ---------------------------------------------------------------------------
// Generic M=32 GEMM: C[32,N] = A[32,K](bf16 row-major) x B(bf16, stored as
// Bt[n*K + k], i.e. one output-column per row, contiguous in k).
// One wave computes one 16x16 tile via v_wmma_f32_16x16x32_bf16.
// A-fragment layout (16-bit A 16x32): lane m=lane&15, half=lane>>4,
//   vector elems 0..7  -> k = half*16 + 0..7     (one 16B load)
//   vector elems 8..15 -> k = half*16 + 8..15    (one 16B load)
// B-fragment layout (16-bit B 32x16): lane n=lane&15, koff=(lane>>4)*16,
//   vector elems 0..15 -> k = koff..koff+15      (one 32B load)
// C/D f32 16x16: VGPR v -> row v + (lane>>4)*8, col lane&15.
// ---------------------------------------------------------------------------
template <int K>
__global__ void wmma_gemm32(const bf16* __restrict__ A, const bf16* __restrict__ Bt,
                            float* __restrict__ C, int N) {
  int wid  = (blockIdx.x * blockDim.x + threadIdx.x) >> 5;
  int lane = threadIdx.x & 31;
  int ntiles = N >> 4;
  if (wid >= ntiles * 2) return;             // whole-wave uniform: EXEC stays all-1s
  int ntile = wid % ntiles;
  int mtile = wid / ntiles;
  int half  = lane >> 4;
  const bf16* Arow = A  + ((mtile << 4) + (lane & 15)) * K + half * 16;
  const bf16* Brow = Bt + ((ntile << 4) + (lane & 15)) * K + half * 16;
  v8f acc = {};
#pragma unroll
  for (int k = 0; k < K; k += 32) {
    union { v16bf v; v8bf h[2]; } a;
    a.h[0] = *reinterpret_cast<const v8bf*>(Arow + k);
    a.h[1] = *reinterpret_cast<const v8bf*>(Arow + k + 8);
    v16bf b = *reinterpret_cast<const v16bf*>(Brow + k);
    acc = __builtin_amdgcn_wmma_f32_16x16x32_bf16(false, a.v, false, b,
                                                  (short)0, acc, false, false);
  }
  int n = (ntile << 4) + (lane & 15);
  int mbase = (mtile << 4) + half * 8;
#pragma unroll
  for (int v = 0; v < 8; ++v) C[(mbase + v) * N + n] = acc[v];
}

// ---------------------------------------------------------------------------
// Setup / conversion kernels (run once per launch)
// ---------------------------------------------------------------------------
__global__ void k_zero(float* p, int n) {
  int i = blockIdx.x * blockDim.x + threadIdx.x;
  if (i < n) p[i] = 0.0f;
}

__global__ void k_cvt(const float* __restrict__ s, bf16* __restrict__ d, int n) {
  int i = blockIdx.x * blockDim.x + threadIdx.x;
  if (i < n) d[i] = (bf16)s[i];
}

// dst[n*1024 + k] = (k<512 ? Wih[n,k] : Whh[n,k-512]),  n in [0,2048)
__global__ void k_cvt_wcat(const float* __restrict__ Wih, const float* __restrict__ Whh,
                           bf16* __restrict__ dst) {
  int i = blockIdx.x * blockDim.x + threadIdx.x;
  if (i >= 2048 * 1024) return;
  int n = i >> 10, k = i & 1023;
  float v = (k < 512) ? Wih[n * 512 + k] : Whh[n * 512 + (k - 512)];
  dst[i] = (bf16)v;
}

// transpose: src[R,Ccols] -> dst[Ccols,R]
__global__ void k_cvt_T(const float* __restrict__ src, bf16* __restrict__ dst,
                        int R, int Ccols) {
  int i = blockIdx.x * blockDim.x + threadIdx.x;
  if (i >= R * Ccols) return;
  int r = i / Ccols, c = i % Ccols;
  dst[c * R + r] = (bf16)src[i];
}

__global__ void k_init(const float* __restrict__ h0, const float* __restrict__ c0,
                       const float* __restrict__ prev0,
                       bf16* __restrict__ A0, bf16* __restrict__ A1,
                       float* __restrict__ c_state, float* __restrict__ prev_out) {
  int i = blockIdx.x * blockDim.x + threadIdx.x;
  if (i < 2 * B_ * H_) {
    int l = i / (B_ * H_), r = i % (B_ * H_);
    int b = r / H_, j = r % H_;
    bf16 hv = (bf16)h0[i];
    if (l == 0) A0[b * 1024 + 512 + j] = hv;
    else        A1[b * 1024 + 512 + j] = hv;
    c_state[i] = c0[i];
  }
  if (i < B_ * E_) prev_out[i] = prev0[i];
}

// ---------------------------------------------------------------------------
// Per-step kernels
// ---------------------------------------------------------------------------
__global__ void k_embed(const int* __restrict__ abstract, const float* __restrict__ embW,
                        const float* __restrict__ prev_out, const float* __restrict__ v_i,
                        bf16* __restrict__ A0, float* __restrict__ gate_emb, int t) {
  int b = blockIdx.x, e = threadIdx.x;                 // 256 threads
  int tok = abstract[b * T_ + t];
  float emb = embW[(size_t)tok * E_ + e];
  A0[b * 1024 + e]       = (bf16)emb;
  A0[b * 1024 + 256 + e] = (bf16)prev_out[b * E_ + e];
  __shared__ float red[256];
  red[e] = emb * v_i[e];
  __syncthreads();
  for (int s = 128; s > 0; s >>= 1) { if (e < s) red[e] += red[e + s]; __syncthreads(); }
  if (e == 0) gate_emb[b] = red[0];
}

// LSTM pointwise cell; writes new h as bf16 into up to 3 destination A-buffers.
__global__ void k_cell(const float* __restrict__ g, const float* __restrict__ b_ih,
                       const float* __restrict__ b_hh, float* __restrict__ c_state,
                       bf16* d0, int s0, int o0,
                       bf16* d1, int s1, int o1,
                       bf16* d2, int s2, int o2,
                       float* hf32) {
  int i = blockIdx.x * blockDim.x + threadIdx.x;       // B*H = 16384
  int b = i / H_, j = i % H_;
  const float* gb = g + b * 4 * H_;
  float gi = gb[j]            + b_ih[j]            + b_hh[j];
  float gf = gb[H_ + j]       + b_ih[H_ + j]       + b_hh[H_ + j];
  float gg = gb[2 * H_ + j]   + b_ih[2 * H_ + j]   + b_hh[2 * H_ + j];
  float go = gb[3 * H_ + j]   + b_ih[3 * H_ + j]   + b_hh[3 * H_ + j];
  float c = sigf(gf) * c_state[i] + sigf(gi) * tanhf(gg);
  float h = sigf(go) * tanhf(c);
  c_state[i] = c;
  bf16 hb = (bf16)h;
  if (d0) d0[b * s0 + o0 + j] = hb;
  if (d1) d1[b * s1 + o1 + j] = hb;
  if (d2) d2[b * s2 + o2 + j] = hb;
  if (hf32) hf32[i] = h;
}

// score = query . enc_proj, mask, softmax over L; one block per batch row.
__global__ void k_attn(const float* __restrict__ query, const float* __restrict__ enc_proj,
                       const unsigned char* __restrict__ mask, float* __restrict__ attn) {
  int b = blockIdx.x, tid = threadIdx.x;               // 256 threads
  __shared__ float qs[H_];
  qs[tid] = query[b * H_ + tid];
  qs[tid + 256] = query[b * H_ + tid + 256];
  __syncthreads();
  float sc[2];
#pragma unroll
  for (int p = 0; p < 2; ++p) {
    int l = tid + p * 256;
    const float4* ep = reinterpret_cast<const float4*>(enc_proj + ((size_t)b * L_ + l) * H_);
    float s = 0.0f;
    for (int d = 0; d < H_ / 4; ++d) {
      float4 q4 = reinterpret_cast<const float4*>(qs)[d];
      float4 e4 = ep[d];
      s += q4.x * e4.x + q4.y * e4.y + q4.z * e4.z + q4.w * e4.w;
    }
    sc[p] = mask[b * L_ + l] ? s : -1e9f;
  }
  __shared__ float red[256];
  red[tid] = fmaxf(sc[0], sc[1]);
  __syncthreads();
  for (int s = 128; s > 0; s >>= 1) { if (tid < s) red[tid] = fmaxf(red[tid], red[tid + s]); __syncthreads(); }
  float m = red[0];
  __syncthreads();
  float e0 = __expf(sc[0] - m), e1 = __expf(sc[1] - m);
  red[tid] = e0 + e1;
  __syncthreads();
  for (int s = 128; s > 0; s >>= 1) { if (tid < s) red[tid] += red[tid + s]; __syncthreads(); }
  float inv = 1.0f / red[0];
  attn[b * L_ + tid]       = e0 * inv;
  attn[b * L_ + tid + 256] = e1 * inv;
}

// ctx = attn . enc_mem ; gate = sigmoid(ctx.v_c + h1.v_s + emb.v_i + b);
// scatter attn*gate into `add` at extended-vocab indices. One block per b.
__global__ void k_ctx(const float* __restrict__ attn, const float* __restrict__ enc_mem,
                      const float* __restrict__ h1, const float* __restrict__ v_c,
                      const float* __restrict__ v_s, const float* __restrict__ gate_emb,
                      const float* __restrict__ copy_b, const int* __restrict__ extend_art,
                      bf16* __restrict__ Adec, float* __restrict__ gate_out,
                      float* __restrict__ add) {
  int b = blockIdx.x, tid = threadIdx.x;               // 256 threads
  __shared__ float as[L_];
  as[tid]       = attn[b * L_ + tid];
  as[tid + 256] = attn[b * L_ + tid + 256];
  __syncthreads();
  float acc0 = 0.0f, acc1 = 0.0f;
  for (int l = 0; l < L_; ++l) {
    const float* em = enc_mem + ((size_t)b * L_ + l) * H_;
    float a = as[l];
    acc0 += a * em[tid];
    acc1 += a * em[tid + 256];
  }
  Adec[b * 1024 + 512 + tid]       = (bf16)acc0;
  Adec[b * 1024 + 512 + tid + 256] = (bf16)acc1;
  __shared__ float red[256];
  red[tid] = acc0 * v_c[tid] + acc1 * v_c[tid + 256] +
             h1[b * H_ + tid] * v_s[tid] + h1[b * H_ + tid + 256] * v_s[tid + 256];
  __syncthreads();
  for (int s = 128; s > 0; s >>= 1) { if (tid < s) red[tid] += red[tid + s]; __syncthreads(); }
  __shared__ float gsh;
  if (tid == 0) {
    gsh = sigf(red[0] + gate_emb[b] + copy_b[0]);
    gate_out[b] = gsh;
  }
  __syncthreads();
  float g = gsh;
#pragma unroll
  for (int p = 0; p < 2; ++p) {
    int l = tid + p * 256;
    int idx = extend_art[b * L_ + l];
    idx = idx < 0 ? 0 : (idx > EXT_ - 1 ? EXT_ - 1 : idx);
    atomicAdd(&add[(size_t)b * EXT_ + idx], as[l] * g);
  }
}

__global__ void k_decbias(const float* __restrict__ dec_raw, const float* __restrict__ proj_b,
                          float* __restrict__ prev_out, bf16* __restrict__ Alog) {
  int i = blockIdx.x * blockDim.x + threadIdx.x;       // B*E = 8192
  float v = dec_raw[i] + proj_b[i % E_];
  prev_out[i] = v;
  Alog[i] = (bf16)v;
}

__global__ void k_rowred(const float* __restrict__ logits, float* __restrict__ rowmax,
                         float* __restrict__ rowsum) {
  int b = blockIdx.x, tid = threadIdx.x;               // 256 threads
  float m = -3.4e38f;
  for (int v = tid; v < V_; v += 256) m = fmaxf(m, logits[(size_t)b * V_ + v]);
  __shared__ float red[256];
  red[tid] = m;
  __syncthreads();
  for (int s = 128; s > 0; s >>= 1) { if (tid < s) red[tid] = fmaxf(red[tid], red[tid + s]); __syncthreads(); }
  m = red[0];
  __syncthreads();
  float su = 0.0f;
  for (int v = tid; v < V_; v += 256) su += __expf(logits[(size_t)b * V_ + v] - m);
  red[tid] = su;
  __syncthreads();
  for (int s = 128; s > 0; s >>= 1) { if (tid < s) red[tid] += red[tid + s]; __syncthreads(); }
  if (tid == 0) { rowmax[b] = m; rowsum[b] = red[0]; }
}

__global__ void k_final(const float* __restrict__ logits, const float* __restrict__ rowmax,
                        const float* __restrict__ rowsum, const float* __restrict__ gate,
                        float* __restrict__ add, float* __restrict__ out, int t) {
  int v = blockIdx.x * blockDim.x + threadIdx.x;
  int b = blockIdx.y;
  if (v >= EXT_) return;
  float g = gate[b];
  float p = (v < V_) ? __expf(logits[(size_t)b * V_ + v] - rowmax[b]) / rowsum[b] : 0.0f;
  size_t ai = (size_t)b * EXT_ + v;
  float a = add[ai];
  add[ai] = 0.0f;                                      // re-zero for next step
  out[((size_t)b * T_ + t) * EXT_ + v] = __logf((1.0f - g) * p + a + 1e-12f);
}

// ---------------------------------------------------------------------------
extern "C" void kernel_launch(void* const* d_in, const int* in_sizes, int n_in,
                              void* d_out, int out_size, void* d_ws, size_t ws_size,
                              hipStream_t stream) {
  (void)in_sizes; (void)n_in; (void)out_size; (void)ws_size;
  const int*   abstract = (const int*)  d_in[0];
  const float* enc_mem  = (const float*)d_in[1];
  const float* enc_proj = (const float*)d_in[2];
  const unsigned char* mask = (const unsigned char*)d_in[3];
  const int*   ext_art  = (const int*)  d_in[4];
  const float* h0       = (const float*)d_in[5];
  const float* c0       = (const float*)d_in[6];
  const float* prev0    = (const float*)d_in[7];
  const float* embW     = (const float*)d_in[8];
  const float* Wih0     = (const float*)d_in[9];
  const float* Whh0     = (const float*)d_in[10];
  const float* bih0     = (const float*)d_in[11];
  const float* bhh0     = (const float*)d_in[12];
  const float* Wih1     = (const float*)d_in[13];
  const float* Whh1     = (const float*)d_in[14];
  const float* bih1     = (const float*)d_in[15];
  const float* bhh1     = (const float*)d_in[16];
  const float* attnW    = (const float*)d_in[17];
  const float* projW    = (const float*)d_in[18];
  const float* projb    = (const float*)d_in[19];
  const float* v_c      = (const float*)d_in[20];
  const float* v_s      = (const float*)d_in[21];
  const float* v_i      = (const float*)d_in[22];
  const float* copy_b   = (const float*)d_in[23];
  float* out = (float*)d_out;

  char* ws = (char*)d_ws;
  size_t off = 0;
  auto alloc = [&](size_t bytes) { size_t o = off; off += (bytes + 255) & ~(size_t)255; return o; };
  bf16* embWb   = (bf16*)(ws + alloc((size_t)V_ * E_ * 2));
  bf16* wcat0   = (bf16*)(ws + alloc((size_t)2048 * 1024 * 2));
  bf16* wcat1   = (bf16*)(ws + alloc((size_t)2048 * 1024 * 2));
  bf16* attnT   = (bf16*)(ws + alloc((size_t)512 * 512 * 2));
  bf16* projT   = (bf16*)(ws + alloc((size_t)1024 * 256 * 2));
  bf16* A0      = (bf16*)(ws + alloc((size_t)B_ * 1024 * 2));
  bf16* A1      = (bf16*)(ws + alloc((size_t)B_ * 1024 * 2));
  bf16* Aq      = (bf16*)(ws + alloc((size_t)B_ * 512 * 2));
  bf16* Adec    = (bf16*)(ws + alloc((size_t)B_ * 1024 * 2));
  bf16* Alog    = (bf16*)(ws + alloc((size_t)B_ * 256 * 2));
  float* g0     = (float*)(ws + alloc((size_t)B_ * 2048 * 4));
  float* g1     = (float*)(ws + alloc((size_t)B_ * 2048 * 4));
  float* cstate = (float*)(ws + alloc((size_t)2 * B_ * H_ * 4));
  float* h1f    = (float*)(ws + alloc((size_t)B_ * H_ * 4));
  float* query  = (float*)(ws + alloc((size_t)B_ * H_ * 4));
  float* attn   = (float*)(ws + alloc((size_t)B_ * L_ * 4));
  float* prevo  = (float*)(ws + alloc((size_t)B_ * E_ * 4));
  float* decraw = (float*)(ws + alloc((size_t)B_ * E_ * 4));
  float* gemb   = (float*)(ws + alloc((size_t)B_ * 4));
  float* gate   = (float*)(ws + alloc((size_t)B_ * 4));
  float* logits = (float*)(ws + alloc((size_t)B_ * V_ * 4));
  float* add    = (float*)(ws + alloc((size_t)B_ * EXT_ * 4));
  float* rowmax = (float*)(ws + alloc((size_t)B_ * 4));
  float* rowsum = (float*)(ws + alloc((size_t)B_ * 4));

  // ---- per-launch setup ----
  k_zero<<<(B_ * EXT_ + 255) / 256, 256, 0, stream>>>(add, B_ * EXT_);
  k_cvt<<<((size_t)V_ * E_ + 255) / 256, 256, 0, stream>>>(embW, embWb, V_ * E_);
  k_cvt_wcat<<<(2048 * 1024) / 256, 256, 0, stream>>>(Wih0, Whh0, wcat0);
  k_cvt_wcat<<<(2048 * 1024) / 256, 256, 0, stream>>>(Wih1, Whh1, wcat1);
  k_cvt_T<<<(512 * 512) / 256, 256, 0, stream>>>(attnW, attnT, 512, 512);
  k_cvt_T<<<(1024 * 256) / 256, 256, 0, stream>>>(projW, projT, 1024, 256);
  k_init<<<(2 * B_ * H_) / 256, 256, 0, stream>>>(h0, c0, prev0, A0, A1, cstate, prevo);

  // ---- 64 sequential decode steps ----
  for (int t = 0; t < T_; ++t) {
    k_embed<<<B_, 256, 0, stream>>>(abstract, embW, prevo, v_i, A0, gemb, t);
    // gates0 = [x | h0] @ [Wih0 | Whh0]^T       (M=32, K=1024, N=2048)
    wmma_gemm32<1024><<<32, 256, 0, stream>>>(A0, wcat0, g0, 2048);
    k_cell<<<(B_ * H_) / 256, 256, 0, stream>>>(g0, bih0, bhh0, cstate,
                                                A1, 1024, 0,      // h0n -> gates1 input
                                                A0, 1024, 512,    // h0n -> next-step gates0
                                                nullptr, 0, 0, nullptr);
    // gates1 = [h0n | h1] @ [Wih1 | Whh1]^T     (M=32, K=1024, N=2048)
    wmma_gemm32<1024><<<32, 256, 0, stream>>>(A1, wcat1, g1, 2048);
    k_cell<<<(B_ * H_) / 256, 256, 0, stream>>>(g1, bih1, bhh1, cstate + B_ * H_,
                                                Aq,   512, 0,     // h1n -> query GEMM
                                                A1,  1024, 512,   // h1n -> next-step gates1
                                                Adec, 1024, 0,    // h1n -> dec GEMM
                                                h1f);
    // query = h1n @ attn_W                      (M=32, K=512, N=512)
    wmma_gemm32<512><<<8, 256, 0, stream>>>(Aq, attnT, query, 512);
    k_attn<<<B_, 256, 0, stream>>>(query, enc_proj, mask, attn);
    k_ctx<<<B_, 256, 0, stream>>>(attn, enc_mem, h1f, v_c, v_s, gemb, copy_b, ext_art,
                                  Adec, gate, add);
    // dec = [h1n | ctx] @ proj_W                (M=32, K=1024, N=256)
    wmma_gemm32<1024><<<4, 256, 0, stream>>>(Adec, projT, decraw, 256);
    k_decbias<<<(B_ * E_) / 256, 256, 0, stream>>>(decraw, projb, prevo, Alog);
    // logits = dec_out @ emb_W^T                (M=32, K=256, N=32000) — dominant GEMM
    wmma_gemm32<256><<<500, 256, 0, stream>>>(Alog, embWb, logits, V_);
    k_rowred<<<B_, 256, 0, stream>>>(logits, rowmax, rowsum);
    dim3 fg((EXT_ + 255) / 256, B_);
    k_final<<<fg, 256, 0, stream>>>(logits, rowmax, rowsum, gate, add, out, t);
  }
}
